// CrossAttention_80161269613034
// MI455X (gfx1250) — compile-verified, matrix operand order checked by reading, not run on hardware
//
#include <hip/hip_runtime.h>
#include <hip/hip_bf16.h>
#include <math.h>

// ---------------------------------------------------------------------------
// CDNA5 (gfx1250) cross-attention, bf16 WMMA everywhere.
// Dominant cost: K/V projections (36864x2048x2048 x2 = 0.62 TFLOP) -> compute
// bound at bf16 WMMA rates, so all GEMMs use V_WMMA_F32_16X16X32_BF16.
// Round 2: software-pipeline the heavy GEMM through registers so global
// loads for tile k+1 overlap the 16 WMMAs of tile k.
// ---------------------------------------------------------------------------

typedef __bf16 bf16;
typedef __attribute__((ext_vector_type(16))) __bf16 bf16x16;
typedef __attribute__((ext_vector_type(8)))  float  f32x8;

static __device__ inline bf16 f2bf(float f) {
    union { float f; unsigned u; } in; in.f = f;
    unsigned u = in.u + (0x7FFFu + ((in.u >> 16) & 1u));   // round-nearest-even
    union { unsigned short s; bf16 h; } out;
    out.s = (unsigned short)(u >> 16);
    return out.h;
}

// 32B fragment load (16 bf16) from a 16B-aligned pointer (LDS or global).
static __device__ inline bf16x16 ld_frag(const bf16* p) {
    union { bf16x16 v; uint4 q[2]; } r;
    r.q[0] = ((const uint4*)p)[0];
    r.q[1] = ((const uint4*)p)[1];
    return r.v;
}

static __device__ inline f32x8 wmma_bf16(bf16x16 a, bf16x16 b, f32x8 c) {
    return __builtin_amdgcn_wmma_f32_16x16x32_bf16(
        /*neg_a=*/false, a, /*neg_b=*/false, b,
        /*c_mod=*/(short)0, c, /*reuse_a=*/false, /*reuse_b=*/false);
}

// ---------------------------------------------------------------------------
// Weight transpose + fp32->bf16 convert: W[in][out] -> WT[out][in]
// ---------------------------------------------------------------------------
__global__ void __launch_bounds__(256)
ca_wt_conv(const float* __restrict__ W, bf16* __restrict__ WT,
           int rowsIn, int colsOut) {
    __shared__ float t[32][33];
    const int ox = blockIdx.x * 32, iy = blockIdx.y * 32;
    const int tx = threadIdx.x & 31, ty = threadIdx.x >> 5;  // 32x8
    #pragma unroll
    for (int i = 0; i < 32; i += 8)
        t[ty + i][tx] = W[(size_t)(iy + ty + i) * colsOut + ox + tx];
    __syncthreads();
    #pragma unroll
    for (int i = 0; i < 32; i += 8)
        WT[(size_t)(ox + ty + i) * rowsIn + iy + tx] = f2bf(t[tx][ty + i]);
}

// part [64][8][1024] f32 -> Pb [64][16][1024] bf16, rows 8..15 zeroed.
__global__ void __launch_bounds__(256)
ca_part_conv(const float* __restrict__ part, bf16* __restrict__ Pb) {
    const int idx = blockIdx.x * 256 + threadIdx.x;      // < 1024*1024
    const int b = idx >> 14, rem = idx & 16383;
    const int m = rem >> 10, k = rem & 1023;
    float v = (m < 8) ? part[((size_t)b * 8 + m) * 1024 + k] : 0.f;
    Pb[idx] = f2bf(v);
}

// ---------------------------------------------------------------------------
// Fused K/V projection GEMM (register-pipelined).
//   A = feature[b, d, n] (fp32, transposed+converted through LDS)
//   Bk/Bv = WkT/WvT [2048][2048] bf16 row-major (K-contiguous)
//   Outputs: Kc[b][n][d] bf16 row-major (via LDS transpose epilogue),
//            Vt[b][j][n] bf16 (direct fragment stores; lane-contiguous in n)
// Tile: 64 (n) x 128 (j) per WG, KB=64, 8 waves in a 2x4 grid (32x32 each).
// ---------------------------------------------------------------------------
__global__ void __launch_bounds__(256)
ca_kv_gemm(const float* __restrict__ feature,
           const bf16* __restrict__ WkT, const bf16* __restrict__ WvT,
           const float* __restrict__ bk, const float* __restrict__ bv,
           bf16* __restrict__ Kc, bf16* __restrict__ Vt) {
    __shared__ __align__(16) bf16 smem[64 * 72 + 2 * 128 * 72];  // 46 KB
    bf16* As = smem;               // [64][72]  (padded stride: 36 banks, cf-free)
    bf16* Bk = smem + 64 * 72;     // [128][72]
    bf16* Bv = Bk + 128 * 72;      // [128][72]

    const int tid = threadIdx.x, w = tid >> 5, lane = tid & 31;
    const int wm = w >> 2, wn = w & 3;           // 2 x 4 wave grid
    const int lm = lane & 15, lh = lane >> 4;
    const int b = blockIdx.z;
    const int n0 = blockIdx.x * 64;              // 576 = 9 * 64 exact
    const int j0 = blockIdx.y * 128;             // 2048 = 16 * 128 exact

    // per-thread load/store roles (fixed across iterations)
    const int am = tid & 63, akb = tid >> 6;     // A: m, k-base (0..3)
    const int bj = tid >> 1, bseg = tid & 1;     // B: row, 32-half segment

    f32x8 accK[2][2] = {}, accV[2][2] = {};

    float aReg[16];
    uint4 bkReg[4], bvReg[4];

    auto load_tiles = [&](int kt) {
        const float* fp = feature + ((size_t)b * 2048 + kt + akb) * 576 + n0 + am;
        #pragma unroll
        for (int i = 0; i < 16; ++i) aReg[i] = fp[(size_t)(4 * i) * 576];
        const uint4* gk = (const uint4*)(WkT + (size_t)(j0 + bj) * 2048 + kt + bseg * 32);
        const uint4* gv = (const uint4*)(WvT + (size_t)(j0 + bj) * 2048 + kt + bseg * 32);
        #pragma unroll
        for (int i = 0; i < 4; ++i) { bkReg[i] = gk[i]; bvReg[i] = gv[i]; }
    };
    auto store_tiles = [&]() {
        #pragma unroll
        for (int i = 0; i < 16; ++i) As[am * 72 + akb + 4 * i] = f2bf(aReg[i]);
        uint4* dk = (uint4*)(Bk + bj * 72 + bseg * 32);
        uint4* dv = (uint4*)(Bv + bj * 72 + bseg * 32);
        #pragma unroll
        for (int i = 0; i < 4; ++i) { dk[i] = bkReg[i]; dv[i] = bvReg[i]; }
    };

    load_tiles(0);                                // prologue fetch
    for (int it = 0; it < 32; ++it) {
        __syncthreads();                          // LDS free from previous compute
        store_tiles();
        __syncthreads();
        if (it + 1 < 32) load_tiles((it + 1) * 64);   // overlap with WMMAs below
        #pragma unroll
        for (int ks = 0; ks < 2; ++ks) {
            bf16x16 a[2], fk[2], fv[2];
            #pragma unroll
            for (int mf = 0; mf < 2; ++mf)
                a[mf] = ld_frag(As + (wm * 32 + mf * 16 + lm) * 72 + ks * 32 + lh * 16);
            #pragma unroll
            for (int nf = 0; nf < 2; ++nf) {
                fk[nf] = ld_frag(Bk + (wn * 32 + nf * 16 + lm) * 72 + ks * 32 + lh * 16);
                fv[nf] = ld_frag(Bv + (wn * 32 + nf * 16 + lm) * 72 + ks * 32 + lh * 16);
            }
            #pragma unroll
            for (int mf = 0; mf < 2; ++mf)
                #pragma unroll
                for (int nf = 0; nf < 2; ++nf) {
                    accK[mf][nf] = wmma_bf16(a[mf], fk[nf], accK[mf][nf]);
                    accV[mf][nf] = wmma_bf16(a[mf], fv[nf], accV[mf][nf]);
                }
        }
    }
    __syncthreads();

    // V epilogue: D-frag lane holds 8 consecutive n for one column j -> b128 store
    #pragma unroll
    for (int mf = 0; mf < 2; ++mf)
        #pragma unroll
        for (int nf = 0; nf < 2; ++nf) {
            const int j  = j0 + wn * 32 + nf * 16 + lm;
            const int nb = n0 + wm * 32 + mf * 16 + lh * 8;
            const float bias = bv[j];
            union { bf16 h[8]; uint4 q; } pk;
            #pragma unroll
            for (int v = 0; v < 8; ++v) pk.h[v] = f2bf(accV[mf][nf][v] + bias);
            *(uint4*)(Vt + ((size_t)b * 2048 + j) * 576 + nb) = pk.q;
        }

    // K epilogue: transpose through LDS, then coalesced row-major stores
    bf16* Ks = smem;  // reuse (all compute reads done at last barrier)
    #pragma unroll
    for (int mf = 0; mf < 2; ++mf)
        #pragma unroll
        for (int nf = 0; nf < 2; ++nf) {
            const int col = wn * 32 + nf * 16 + lm;
            const float bias = bk[j0 + col];
            #pragma unroll
            for (int v = 0; v < 8; ++v)
                Ks[(wm * 32 + mf * 16 + v + lh * 8) * 136 + col] =
                    f2bf(accK[mf][nf][v] + bias);
        }
    __syncthreads();
    for (int c = tid; c < 64 * 16; c += 256) {
        const int row = c >> 4, seg = c & 15;
        uint4 val = *(uint4*)(Ks + row * 136 + seg * 8);
        *(uint4*)(Kc + ((size_t)b * 576 + n0 + row) * 2048 + j0 + seg * 8) = val;
    }
}

// ---------------------------------------------------------------------------
// Generic GEMM over 1024 packed batch rows (64 batches x 16 padded rows).
//   C[1024][2048] = A[1024][Kdim] (bf16) x BT[2048][Kdim]^T (bf16) + bias
// Tile 64x128 per WG; B stays L2-resident (weights read 16x from L2 only).
// mode 0: store bf16 (val*scale)      [Q projection]
// mode 1: store f32                   [MLP hidden, pre-LN]
// mode 2: +resid f32, store bf16      [MLP out + residual]
// mode 3: store f32 rows k<8 packed to d_out [output projection]
// ---------------------------------------------------------------------------
__global__ void __launch_bounds__(256)
ca_gemm_rows(const bf16* __restrict__ A, int lda,
             const bf16* __restrict__ BT, int ldb,
             const float* __restrict__ bias, int Kdim,
             float scale, int mode,
             const float* __restrict__ resid,
             float* __restrict__ outF, bf16* __restrict__ outB) {
    const int tid = threadIdx.x, w = tid >> 5, lane = tid & 31;
    const int wm = w >> 2, wn = w & 3;
    const int lm = lane & 15, lh = lane >> 4;
    const int r_base = blockIdx.x * 64 + wm * 32;
    const int j_base = blockIdx.y * 128 + wn * 32;

    f32x8 acc[2][2] = {};
    #pragma unroll 2
    for (int kt = 0; kt < Kdim; kt += 32) {
        bf16x16 a[2], fb[2];
        #pragma unroll
        for (int mf = 0; mf < 2; ++mf)
            a[mf] = ld_frag(A + (size_t)(r_base + mf * 16 + lm) * lda + kt + lh * 16);
        #pragma unroll
        for (int nf = 0; nf < 2; ++nf)
            fb[nf] = ld_frag(BT + (size_t)(j_base + nf * 16 + lm) * ldb + kt + lh * 16);
        #pragma unroll
        for (int mf = 0; mf < 2; ++mf)
            #pragma unroll
            for (int nf = 0; nf < 2; ++nf)
                acc[mf][nf] = wmma_bf16(a[mf], fb[nf], acc[mf][nf]);
    }
    #pragma unroll
    for (int mf = 0; mf < 2; ++mf)
        #pragma unroll
        for (int nf = 0; nf < 2; ++nf) {
            const int j = j_base + nf * 16 + lm;
            const float bs = bias ? bias[j] : 0.f;
            #pragma unroll
            for (int v = 0; v < 8; ++v) {
                const int r = r_base + mf * 16 + v + lh * 8;
                float val = acc[mf][nf][v] + bs;
                if (mode == 0) {
                    outB[(size_t)r * 2048 + j] = f2bf(val * scale);
                } else if (mode == 1) {
                    outF[(size_t)r * 2048 + j] = val;
                } else if (mode == 2) {
                    val += resid[(size_t)r * 2048 + j];
                    outB[(size_t)r * 2048 + j] = f2bf(val);
                } else {
                    const int bb = r >> 4, k = r & 15;
                    if (k < 8) outF[((size_t)(bb * 8 + k)) * 2048 + j] = val;
                }
            }
        }
}

// ---------------------------------------------------------------------------
// scores = Qb[b] (16x2048, 1/sqrt(D) pre-folded) x Kc[b]^T -> softmax over 576
// Writes A_v fp32 to d_out and bf16 (rows 8..15 zero) for the F_p GEMM.
// ---------------------------------------------------------------------------
__global__ void __launch_bounds__(256)
ca_attn_scores(const bf16* __restrict__ Qb, const bf16* __restrict__ Kc,
               float* __restrict__ AvOut, bf16* __restrict__ Ab) {
    __shared__ float S[16 * 592];            // 37.9 KB
    const int tid = threadIdx.x, w = tid >> 5, lane = tid & 31;
    const int lm = lane & 15, lh = lane >> 4;
    const int b = blockIdx.x;

    f32x8 acc[5] = {};                       // 36 n-frags over 8 waves
    for (int kt = 0; kt < 2048; kt += 32) {
        bf16x16 a = ld_frag(Qb + (size_t)(b * 16 + lm) * 2048 + kt + lh * 16);
        int idx = 0;
        for (int nf = w; nf < 36; nf += 8, ++idx) {
            bf16x16 kb = ld_frag(Kc + ((size_t)b * 576 + nf * 16 + lm) * 2048 + kt + lh * 16);
            acc[idx] = wmma_bf16(a, kb, acc[idx]);
        }
    }
    {
        int idx = 0;
        for (int nf = w; nf < 36; nf += 8, ++idx)
            #pragma unroll
            for (int v = 0; v < 8; ++v)
                S[(v + lh * 8) * 592 + nf * 16 + lm] = acc[idx][v];
    }
    __syncthreads();

    if (w < 8) {                             // wave w handles query row w
        float* row = S + w * 592;
        float mx = -INFINITY;
        for (int n = lane; n < 576; n += 32) mx = fmaxf(mx, row[n]);
        #pragma unroll
        for (int o = 16; o > 0; o >>= 1) mx = fmaxf(mx, __shfl_xor(mx, o, 32));
        float sum = 0.f;
        for (int n = lane; n < 576; n += 32) {
            float e = __expf(row[n] - mx);
            row[n] = e; sum += e;
        }
        #pragma unroll
        for (int o = 16; o > 0; o >>= 1) sum += __shfl_xor(sum, o, 32);
        const float inv = 1.f / sum;
        for (int n = lane; n < 576; n += 32) {
            float p = row[n] * inv;
            AvOut[((size_t)b * 8 + w) * 576 + n] = p;
            Ab[((size_t)b * 16 + w) * 576 + n] = f2bf(p);
        }
    }
    for (int i = tid; i < 8 * 576; i += 256)           // zero padded rows
        Ab[((size_t)b * 16 + 8 + i / 576) * 576 + (i % 576)] = f2bf(0.f);
}

// ---------------------------------------------------------------------------
// F_p = A_v (16x576) x V (576x2048) using Vt[b][j][n]; store fp32 + bf16.
// ---------------------------------------------------------------------------
__global__ void __launch_bounds__(256)
ca_fp_gemm(const bf16* __restrict__ Ab, const bf16* __restrict__ Vt,
           float* __restrict__ Fp32, bf16* __restrict__ Fpb) {
    const int tid = threadIdx.x, w = tid >> 5, lane = tid & 31;
    const int lm = lane & 15, lh = lane >> 4;
    const int b = blockIdx.x;

    f32x8 acc[16] = {};
    #pragma unroll 2
    for (int kt = 0; kt < 576; kt += 32) {
        bf16x16 a = ld_frag(Ab + (size_t)(b * 16 + lm) * 576 + kt + lh * 16);
        #pragma unroll
        for (int nf = 0; nf < 16; ++nf) {
            const int j = w * 256 + nf * 16 + lm;
            bf16x16 vb = ld_frag(Vt + ((size_t)b * 2048 + j) * 576 + kt + lh * 16);
            acc[nf] = wmma_bf16(a, vb, acc[nf]);
        }
    }
    #pragma unroll
    for (int nf = 0; nf < 16; ++nf) {
        const int j = w * 256 + nf * 16 + lm;
        #pragma unroll
        for (int v = 0; v < 8; ++v) {
            const int m = v + lh * 8;
            const size_t o = ((size_t)b * 16 + m) * 2048 + j;
            Fp32[o] = acc[nf][v];
            Fpb[o]  = f2bf(acc[nf][v]);
        }
    }
}

// ---------------------------------------------------------------------------
// Per-row LayerNorm + ReLU -> bf16 (padded rows -> zero).
// ---------------------------------------------------------------------------
__global__ void __launch_bounds__(256)
ca_ln_relu(const float* __restrict__ H, const float* __restrict__ g,
           const float* __restrict__ be, bf16* __restrict__ A2b) {
    const int r = blockIdx.x;
    __shared__ float red[16];
    if ((r & 15) >= 8) {
        for (int i = threadIdx.x; i < 2048; i += 256)
            A2b[(size_t)r * 2048 + i] = f2bf(0.f);
        return;
    }
    const float* x = H + (size_t)r * 2048;
    float s = 0.f, ss = 0.f;
    for (int i = threadIdx.x; i < 2048; i += 256) {
        float v = x[i]; s += v; ss += v * v;
    }
    #pragma unroll
    for (int o = 16; o > 0; o >>= 1) {
        s += __shfl_xor(s, o, 32); ss += __shfl_xor(ss, o, 32);
    }
    const int w = threadIdx.x >> 5, lane = threadIdx.x & 31;
    if (lane == 0) { red[w] = s; red[w + 8] = ss; }
    __syncthreads();
    float S = 0.f, SS = 0.f;
    #pragma unroll
    for (int i = 0; i < 8; ++i) { S += red[i]; SS += red[i + 8]; }
    const float mu = S * (1.f / 2048.f);
    const float var = SS * (1.f / 2048.f) - mu * mu;
    const float inv = rsqrtf(var + 1e-5f);
    for (int i = threadIdx.x; i < 2048; i += 256) {
        float v = (x[i] - mu) * inv * g[i] + be[i];
        A2b[(size_t)r * 2048 + i] = f2bf(fmaxf(v, 0.f));
    }
}

// ---------------------------------------------------------------------------
extern "C" void kernel_launch(void* const* d_in, const int* in_sizes, int n_in,
                              void* d_out, int out_size, void* d_ws, size_t ws_size,
                              hipStream_t stream) {
    const float* feature = (const float*)d_in[0];
    const float* part    = (const float*)d_in[1];
    const float* Wk = (const float*)d_in[2];  const float* bk = (const float*)d_in[3];
    const float* Wq = (const float*)d_in[4];  const float* bq = (const float*)d_in[5];
    const float* Wv = (const float*)d_in[6];  const float* bv = (const float*)d_in[7];
    const float* W1 = (const float*)d_in[8];  const float* b1 = (const float*)d_in[9];
    const float* lng = (const float*)d_in[10]; const float* lnb = (const float*)d_in[11];
    const float* W2 = (const float*)d_in[12]; const float* b2 = (const float*)d_in[13];
    const float* Wp = (const float*)d_in[14]; const float* bp = (const float*)d_in[15];

    float* outFp = (float*)d_out;                    // [64][8][2048]
    float* outAv = outFp + (size_t)64 * 8 * 2048;    // [64][8][576]

    // Workspace layout (~382 MB total)
    bf16* WkT = (bf16*)d_ws;
    bf16* WvT = WkT + (size_t)2048 * 2048;
    bf16* WqT = WvT + (size_t)2048 * 2048;
    bf16* W1T = WqT + (size_t)2048 * 1024;
    bf16* W2T = W1T + (size_t)2048 * 2048;
    bf16* WpT = W2T + (size_t)2048 * 2048;
    bf16* Pb  = WpT + (size_t)2048 * 2048;           // [1024][1024]
    bf16* Kc  = Pb  + (size_t)1024 * 1024;           // [64][576][2048]
    bf16* Vt  = Kc  + (size_t)64 * 576 * 2048;       // [64][2048][576]
    bf16* Qb  = Vt  + (size_t)64 * 2048 * 576;       // [1024][2048]
    bf16* Ab  = Qb  + (size_t)1024 * 2048;           // [1024][576]
    bf16* Fpb = Ab  + (size_t)1024 * 576;            // [1024][2048]
    bf16* A2b = Fpb + (size_t)1024 * 2048;
    bf16* A3b = A2b + (size_t)1024 * 2048;
    float* Fp32 = (float*)(A3b + (size_t)1024 * 2048);
    float* Hbuf = Fp32 + (size_t)1024 * 2048;

    const float qscale = 0.02209708691207961f;       // 1/sqrt(2048)

    // Weight transpose/convert
    ca_wt_conv<<<dim3(64, 64), 256, 0, stream>>>(Wk, WkT, 2048, 2048);
    ca_wt_conv<<<dim3(64, 64), 256, 0, stream>>>(Wv, WvT, 2048, 2048);
    ca_wt_conv<<<dim3(64, 32), 256, 0, stream>>>(Wq, WqT, 1024, 2048);
    ca_wt_conv<<<dim3(64, 64), 256, 0, stream>>>(W1, W1T, 2048, 2048);
    ca_wt_conv<<<dim3(64, 64), 256, 0, stream>>>(W2, W2T, 2048, 2048);
    ca_wt_conv<<<dim3(64, 64), 256, 0, stream>>>(Wp, WpT, 2048, 2048);
    ca_part_conv<<<4096, 256, 0, stream>>>(part, Pb);

    // Heavy fused K/V projection (the 0.62 TFLOP part)
    ca_kv_gemm<<<dim3(9, 16, 64), 256, 0, stream>>>(feature, WkT, WvT, bk, bv, Kc, Vt);

    // Q projection (scale folded)
    ca_gemm_rows<<<dim3(16, 16), 256, 0, stream>>>(Pb, 1024, WqT, 1024, bq, 1024,
                                                   qscale, 0, nullptr, nullptr, Qb);
    // scores + softmax
    ca_attn_scores<<<64, 256, 0, stream>>>(Qb, Kc, outAv, Ab);
    // F_p = A_v @ V
    ca_fp_gemm<<<64, 256, 0, stream>>>(Ab, Vt, Fp32, Fpb);
    // MLP: h = relu(LN(F_p@W1+b1))@W2 + b2 + F_p
    ca_gemm_rows<<<dim3(16, 16), 256, 0, stream>>>(Fpb, 2048, W1T, 2048, b1, 2048,
                                                   1.f, 1, nullptr, Hbuf, nullptr);
    ca_ln_relu<<<1024, 256, 0, stream>>>(Hbuf, lng, lnb, A2b);
    ca_gemm_rows<<<dim3(16, 16), 256, 0, stream>>>(A2b, 2048, W2T, 2048, b2, 2048,
                                                   1.f, 2, Fp32, nullptr, A3b);
    // Output projection -> d_out
    ca_gemm_rows<<<dim3(16, 16), 256, 0, stream>>>(A3b, 2048, WpT, 2048, bp, 2048,
                                                   1.f, 3, nullptr, outFp, nullptr);
}
